// Tacotron_64123861729529
// MI455X (gfx1250) — matrix operand order repaired and checked by reading, not compiled, not run on hardware
//
#include <hip/hip_runtime.h>

// ---------------------------------------------------------------------------
// Tacotron decoder for MI455X (gfx1250): bf16 WMMA GEMMs (f32 accumulate),
// persistent kernel with grid-wide barrier across the T=250 recurrence.
// Gate GEMM -> LSTM elementwise fused per-workgroup through LDS (each block
// owns i/f/g/o tiles for one 32-wide h range); dual WMMA accumulators break
// the XDL RAW chain; energy+softmax fused block-locally.
// ---------------------------------------------------------------------------

typedef __bf16 bf16;
typedef __attribute__((ext_vector_type(16))) __bf16 v16bf;
typedef __attribute__((ext_vector_type(8)))  __bf16 v8bf;
typedef __attribute__((ext_vector_type(8)))  float   v8f;

constexpr int cB = 16, cL = 256, cE = 512, cH = 1024, cP = 256;
constexpr int cOUT = 80, cT = 250, cA = 128, cFLT = 32;
constexpr int cKATT = cP + cE + cH;     // 1792 = [pre | context | h_att]
constexpr int cKGEN = cH + cE + cH;     // 2560 = [h_att | context | h_gen]
constexpr int cNG   = 4 * cH;           // 4096 gate outputs
constexpr int NWG = 32, TPB = 256, NT = NWG * TPB;   // 8192 threads, 256 waves

// ---------------- WMMA fragment helpers (CDNA5 16x16x32 bf16) ---------------

// A: 16x32 bf16 tile from row-major matrix (stride in elements).
// Lane l (l<16): row l, K = {k0..k0+7, k0+16..k0+23}; lane l+16: {+8, +24}.
__device__ __forceinline__ v16bf load_a_frag(const bf16* A, int row0, int k0,
                                             int stride) {
  int lane = threadIdx.x & 31;
  int m = lane & 15, half = lane >> 4;
  const bf16* p = A + (size_t)(row0 + m) * stride + k0 + half * 8;
  v8bf lo = *(const v8bf*)(p);
  v8bf hi = *(const v8bf*)(p + 16);
  v16bf r;
#pragma unroll
  for (int i = 0; i < 8; ++i) { r[i] = lo[i]; r[i + 8] = hi[i]; }
  return r;
}

// B: pre-packed 32x16 (KxN) tile blobs of 512 bf16; lane's 16 values
// contiguous (lane<16: N=lane, K {0..7,16..23}; lane>=16: N=lane-16, K {8..15,24..31}).
__device__ __forceinline__ v16bf load_b_frag(const bf16* Bp, long blob) {
  const bf16* p = Bp + blob * 512 + (size_t)(threadIdx.x & 31) * 16;
  v8bf lo = *(const v8bf*)(p);
  v8bf hi = *(const v8bf*)(p + 8);
  v16bf r;
#pragma unroll
  for (int i = 0; i < 8; ++i) { r[i] = lo[i]; r[i + 8] = hi[i]; }
  return r;
}

// Two independent accumulators over even/odd K tiles: consecutive v_wmma ops
// are independent, avoiding the WMMA->WMMA RAW hazard (~5 slots for bf16).
__device__ __forceinline__ v8f gemm_tile(const bf16* A, int aStride, int row0,
                                         const bf16* Bp, int nt, int kTiles) {
  long base = (long)nt * kTiles;
  v8f c0 = {}, c1 = {};
  int kt = 0;
#pragma unroll 2
  for (; kt + 1 < kTiles; kt += 2) {
    v16bf a0 = load_a_frag(A, row0, kt * 32, aStride);
    v16bf b0 = load_b_frag(Bp, base + kt);
    c0 = __builtin_amdgcn_wmma_f32_16x16x32_bf16(false, a0, false, b0,
                                                 (short)0, c0, false, false);
    v16bf a1 = load_a_frag(A, row0, kt * 32 + 32, aStride);
    v16bf b1 = load_b_frag(Bp, base + kt + 1);
    c1 = __builtin_amdgcn_wmma_f32_16x16x32_bf16(false, a1, false, b1,
                                                 (short)0, c1, false, false);
  }
  if (kt < kTiles) {
    v16bf a = load_a_frag(A, row0, kt * 32, aStride);
    v16bf b = load_b_frag(Bp, base + kt);
    c0 = __builtin_amdgcn_wmma_f32_16x16x32_bf16(false, a, false, b,
                                                 (short)0, c0, false, false);
  }
#pragma unroll
  for (int i = 0; i < 8; ++i) c0[i] += c1[i];
  return c0;
}

// C layout: VGPR r holds row (r + 8*(lane>=16)), col = lane&15.
__device__ __forceinline__ void store_c_f32(float* C, long cStride, int row0,
                                            int col0, const v8f& c) {
  int lane = threadIdx.x & 31;
  int n = lane & 15, mb = (lane >> 4) * 8;
#pragma unroll
  for (int r = 0; r < 8; ++r)
    C[(size_t)(row0 + mb + r) * cStride + col0 + n] = c[r];
}

__device__ __forceinline__ float sigm(float x) {
  return 1.0f / (1.0f + __expf(-x));
}

// ------------------------- weight packing (once) ----------------------------

enum { PK_ATT, PK_GEN, PK_WQ, PK_WFS, PK_W1, PK_W2, PK_WMEM };

__global__ void pack_b_kernel(bf16* dst, const float* S0, const float* S1,
                              int which, int kTiles, int nTiles) {
  int total = kTiles * nTiles * 512;
  for (int i = blockIdx.x * blockDim.x + threadIdx.x; i < total;
       i += gridDim.x * blockDim.x) {
    int blob = i >> 9, e = i & 511;
    int nt = blob / kTiles, kt = blob - nt * kTiles;
    int l = e >> 4, w = e & 15;
    int K = kt * 32 + (l >> 4) * 8 + (w & 7) + (w >> 3) * 16;
    int N = nt * 16 + (l & 15);
    float val = 0.0f;
    switch (which) {
      case PK_ATT:  // rows: Wih^T (768) then Whh^T (1024)
        val = (K < 768) ? S0[(size_t)N * 768 + K]
                        : S1[(size_t)N * 1024 + (K - 768)];
        break;
      case PK_GEN:  // rows: Wih^T (1536) then Whh^T (1024)
        val = (K < 1536) ? S0[(size_t)N * 1536 + K]
                         : S1[(size_t)N * 1024 + (K - 1536)];
        break;
      case PK_WQ:   val = S0[(size_t)N * 1024 + K]; break;
      case PK_WFS: {  // A-order is [context(512) | h_gen(1024)]
        int Km = (K < 512) ? (1024 + K) : (K - 512);
        if (N < 80)       val = S0[(size_t)N * 1536 + Km];   // Wframe
        else if (N == 80) val = S1[Km];                      // Wstop
      } break;
      case PK_W1:   val = (K < 80) ? S0[(size_t)N * 80 + K] : 0.0f; break;
      case PK_W2:   val = S0[(size_t)N * 256 + K]; break;
      case PK_WMEM: val = S0[(size_t)N * 512 + K]; break;
    }
    dst[i] = (bf16)val;
  }
}

__global__ void cvt_enc_kernel(bf16* dst, const float* src, int n) {
  for (int i = blockIdx.x * blockDim.x + threadIdx.x; i < n;
       i += gridDim.x * blockDim.x)
    dst[i] = (bf16)src[i];
}

// dec_in[t][b][o<96] = (t>0 && o<80) ? target[b, o, t-1] : 0  (shifted teacher)
__global__ void build_decin_kernel(bf16* dst, const float* target) {
  int total = cT * cB * 96;
  for (int i = blockIdx.x * blockDim.x + threadIdx.x; i < total;
       i += gridDim.x * blockDim.x) {
    int o = i % 96, r = i / 96;
    int b = r % cB, t = r / cB;
    float v = 0.0f;
    if (t > 0 && o < cOUT) v = target[((size_t)b * cOUT + o) * cT + (t - 1)];
    dst[i] = (bf16)v;
  }
}

// Generic prep GEMM: one wave per 16x16 C tile.
__global__ void gemm_prep_kernel(const bf16* A, int aStride, const bf16* Bp,
                                 int mTiles, int nTiles, int kTiles,
                                 const float* bias, int relu, float* Cf,
                                 bf16* Cb, int cStride) {
  int wave = (blockIdx.x * blockDim.x + threadIdx.x) >> 5;
  if (wave >= mTiles * nTiles) return;
  int mt = wave / nTiles, nt = wave - mt * nTiles;
  v8f c = gemm_tile(A, aStride, mt * 16, Bp, nt, kTiles);
  int lane = threadIdx.x & 31;
  int n = lane & 15, mb = (lane >> 4) * 8;
  float bv = bias ? bias[nt * 16 + n] : 0.0f;
#pragma unroll
  for (int r = 0; r < 8; ++r) {
    float v = c[r] + bv;
    if (relu) v = v > 0.0f ? v : 0.0f;
    size_t idx = (size_t)(mt * 16 + mb + r) * cStride + nt * 16 + n;
    if (Cf) Cf[idx] = v;
    if (Cb) Cb[idx] = (bf16)v;
  }
}

// --------------------------- persistent decoder -----------------------------

__device__ __forceinline__ void grid_sync(unsigned* cnt,
                                          volatile unsigned* gen) {
  __syncthreads();
  if (threadIdx.x == 0) {
    __threadfence();
    unsigned g = *gen;
    if (atomicAdd(cnt, 1u) == (unsigned)(NWG - 1)) {
      *cnt = 0u;
      __threadfence();
      *gen = g + 1u;
    } else {
      while (*gen == g) __builtin_amdgcn_s_sleep(2);
    }
    __threadfence();
  }
  __syncthreads();
}

// Fused gate-GEMM + LSTM elementwise. Block bk owns h in [32*bk, 32*bk+32):
// wave lw -> gate (lw>>1), h-subtile (lw&1); tile nt = gate*64 + bk*2 + htile.
// Gates staged in LDS; __syncthreads replaces a grid barrier.
__device__ __forceinline__ void lstm_fused(
    const bf16* x, int kx, const bf16* wp, const float* bias, float* cstate,
    bf16* dst0, int stride0, int off0, bf16* dst1, int stride1, int off1,
    float* sGates) {
  const int bk = blockIdx.x, lw = threadIdx.x >> 5;
  const int gate = lw >> 1, htile = lw & 1;
  const int nt = gate * 64 + bk * 2 + htile;
  v8f c = gemm_tile(x, kx, 0, wp, nt, kx / 32);
  int lane = threadIdx.x & 31;
  int n = lane & 15, mb = (lane >> 4) * 8;
#pragma unroll
  for (int r = 0; r < 8; ++r)
    sGates[((mb + r) * 4 + gate) * 32 + htile * 16 + n] = c[r];
  __syncthreads();
  for (int ii = threadIdx.x; ii < cB * 32; ii += TPB) {
    int b = ii >> 5, hh = ii & 31;
    int h = bk * 32 + hh;
    const float* g = sGates + (size_t)b * 4 * 32 + hh;
    float gi = g[0 * 32] + bias[h];
    float gf = g[1 * 32] + bias[cH + h];
    float gg = g[2 * 32] + bias[2 * cH + h];
    float go = g[3 * 32] + bias[3 * cH + h];
    float c2 = sigm(gf) * cstate[b * cH + h] + sigm(gi) * tanhf(gg);
    float h2 = sigm(go) * tanhf(c2);
    cstate[b * cH + h] = c2;
    bf16 hb = (bf16)h2;
    dst0[(size_t)b * stride0 + off0 + h] = hb;
    if (dst1) dst1[(size_t)b * stride1 + off1 + h] = hb;
  }
  __syncthreads();
}

__global__ void __launch_bounds__(TPB) tacotron_decode_kernel(
    const bf16* __restrict__ wattp, const bf16* __restrict__ wgenp,
    const bf16* __restrict__ wqp, const bf16* __restrict__ wfsp,
    const bf16* __restrict__ pre, const float* __restrict__ memproj,
    const float* __restrict__ enc, const int* __restrict__ enclen,
    const float* __restrict__ att_b, const float* __restrict__ gen_b,
    const float* __restrict__ bframe, const float* __restrict__ bstop,
    const float* __restrict__ wconv, const float* __restrict__ wloc,
    const float* __restrict__ vvec, bf16* xatt, bf16* xgen, float* catt,
    float* cgen, float* qbuf, float* feat, float* prevw, float* cumw,
    unsigned* barcnt, volatile unsigned* bargen, float* out_spec,
    float* out_stop, float* out_align) {
  __shared__ float sGates[cB * 4 * 32];  // [b][gate][h within block range]
  __shared__ float sE[cL];               // energies for one batch row

  const int tid = blockIdx.x * TPB + threadIdx.x;
  const int gwave = tid >> 5;

  // ---- init recurrent state (ws is poisoned; zero everything we carry) ----
  for (int i = tid; i < cB * cKATT; i += NT) xatt[i] = (bf16)0.0f;
  for (int i = tid; i < cB * cKGEN; i += NT) xgen[i] = (bf16)0.0f;
  for (int i = tid; i < cB * cH; i += NT) { catt[i] = 0.0f; cgen[i] = 0.0f; }
  for (int i = tid; i < cB * cL; i += NT) { prevw[i] = 0.0f; cumw[i] = 0.0f; }
  for (int i = tid; i < cB * cP; i += NT) {  // pre[t=0] -> xatt cols 0..255
    int b = i >> 8, p = i & 255;
    xatt[(size_t)b * cKATT + p] = pre[(size_t)b * cP + p];
  }
  grid_sync(barcnt, bargen);

  for (int t = 0; t < cT; ++t) {
    // -- Phase 1: attention-LSTM gates + elementwise (fused via LDS) ---------
    lstm_fused(xatt, cKATT, wattp, att_b, catt,
               xatt, cKATT, cP + cE,        // h_att -> xatt col 768..
               xgen, cKGEN, 0, sGates);     // h_att -> xgen col 0..
    grid_sync(barcnt, bargen);

    // -- Phase 2: q = h_att @ Wq^T (waves 0..7) + location conv (all) --------
    if (gwave < cA / 16) {
      v8f c = gemm_tile(xatt + (cP + cE), cKATT, 0, wqp, gwave, cH / 32);
      store_c_f32(qbuf, cA, 0, gwave * 16, c);
    }
    for (int i = tid; i < cB * cFLT * cL; i += NT) {
      int l = i & 255, f = (i >> 8) & 31, b = i >> 13;
      const float* pw = prevw + b * cL;
      const float* cw = cumw + b * cL;
      const float* k0 = wconv + (size_t)f * 2 * 31;
      const float* k1 = k0 + 31;
      float acc = 0.0f;
#pragma unroll
      for (int j = 0; j < 31; ++j) {
        int ll = l + j - 15;
        if (ll >= 0 && ll < cL) acc += k0[j] * pw[ll] + k1[j] * cw[ll];
      }
      feat[i] = acc;  // [b][f][l]
    }
    grid_sync(barcnt, bargen);

    // -- Phase 3: energy + masked softmax, block-local (block b = row b) -----
    if (blockIdx.x < cB) {
      int b = blockIdx.x, l = threadIdx.x;
      float fv[cFLT];
#pragma unroll
      for (int f = 0; f < cFLT; ++f)
        fv[f] = feat[((size_t)b * cFLT + f) * cL + l];
      const float* qp = qbuf + b * cA;
      const float* mp = memproj + ((size_t)b * cL + l) * cA;
      float e = 0.0f;
      for (int a = 0; a < cA; ++a) {
        float s = qp[a] + mp[a];
        const float* wl = wloc + a * cFLT;
#pragma unroll
        for (int f = 0; f < cFLT; ++f) s += wl[f] * fv[f];
        e += vvec[a] * tanhf(s);
      }
      sE[l] = (l < enclen[b]) ? e : -1.0e9f;
      __syncthreads();
      if (threadIdx.x < 32) {  // wave 0: softmax over 256 via shuffles
        int lane = threadIdx.x;
        float ev[8], m = -3.0e38f;
#pragma unroll
        for (int j = 0; j < 8; ++j) {
          ev[j] = sE[j * 32 + lane];
          m = fmaxf(m, ev[j]);
        }
        for (int o = 16; o; o >>= 1) m = fmaxf(m, __shfl_xor(m, o, 32));
        float s = 0.0f;
#pragma unroll
        for (int j = 0; j < 8; ++j) { ev[j] = __expf(ev[j] - m); s += ev[j]; }
        for (int o = 16; o; o >>= 1) s += __shfl_xor(s, o, 32);
        float inv = 1.0f / s;
#pragma unroll
        for (int j = 0; j < 8; ++j) {
          int ll = j * 32 + lane;
          float w = ev[j] * inv;
          prevw[b * cL + ll] = w;
          cumw[b * cL + ll] += w;
          out_align[((size_t)b * cT + t) * cL + ll] = w;
        }
      }
    }
    grid_sync(barcnt, bargen);

    // -- Phase 4: context[b,e] = sum_l w[b,l]*enc[b,l,e] ---------------------
    for (int i = tid; i < cB * cE; i += NT) {
      int b = i >> 9, e = i & 511;
      const float* w = prevw + b * cL;
      const float* ep = enc + ((size_t)b * cL) * cE + e;
      float acc = 0.0f;
      for (int l = 0; l < cL; ++l) acc += w[l] * ep[(size_t)l * cE];
      bf16 cb = (bf16)acc;
      xatt[(size_t)b * cKATT + cP + e] = cb;
      xgen[(size_t)b * cKGEN + cH + e] = cb;
    }
    grid_sync(barcnt, bargen);

    // -- Phase 5: generator-LSTM gates + elementwise (fused via LDS) ---------
    lstm_fused(xgen, cKGEN, wgenp, gen_b, cgen,
               xgen, cKGEN, cH + cE,        // h_gen -> xgen col 1536..
               nullptr, 0, 0, sGates);
    grid_sync(barcnt, bargen);

    // -- Phase 6: frame/stop GEMM (waves 0..5) + stage pre[t+1] --------------
    if (gwave < 6) {
      v8f c = gemm_tile(xgen + cH, cKGEN, 0, wfsp, gwave, 1536 / 32);
      int lane = threadIdx.x & 31;
      int n = lane & 15, mb = (lane >> 4) * 8;
      int o = gwave * 16 + n;
#pragma unroll
      for (int r = 0; r < 8; ++r) {
        int b = mb + r;
        if (o < cOUT)
          out_spec[((size_t)b * cT + t) * cOUT + o] = c[r] + bframe[o];
        else if (o == cOUT)
          out_stop[(size_t)b * cT + t] = c[r] + bstop[0];
      }
    }
    if (t + 1 < cT) {
      for (int i = tid; i < cB * cP; i += NT) {
        int b = i >> 8, p = i & 255;
        xatt[(size_t)b * cKATT + p] = pre[((size_t)(t + 1) * cB + b) * cP + p];
      }
    }
    grid_sync(barcnt, bargen);
  }
}

// ------------------------------- launcher -----------------------------------

extern "C" void kernel_launch(void* const* d_in, const int* in_sizes, int n_in,
                              void* d_out, int out_size, void* d_ws,
                              size_t ws_size, hipStream_t stream) {
  (void)in_sizes; (void)n_in; (void)out_size; (void)ws_size;
  // Inputs in setup_inputs() dict order (params flattened in insertion order).
  const float* enc     = (const float*)d_in[0];
  const int*   enclen  = (const int*)d_in[1];
  const float* target  = (const float*)d_in[2];
  const float* pre_W1  = (const float*)d_in[4];
  const float* pre_b1  = (const float*)d_in[5];
  const float* pre_W2  = (const float*)d_in[6];
  const float* pre_b2  = (const float*)d_in[7];
  const float* att_Wih = (const float*)d_in[8];
  const float* att_Whh = (const float*)d_in[9];
  const float* att_b   = (const float*)d_in[10];
  const float* gen_Wih = (const float*)d_in[11];
  const float* gen_Whh = (const float*)d_in[12];
  const float* gen_b   = (const float*)d_in[13];
  const float* Wq      = (const float*)d_in[14];
  const float* Wmem    = (const float*)d_in[15];
  const float* Wconv   = (const float*)d_in[16];
  const float* Wloc    = (const float*)d_in[17];
  const float* vvec    = (const float*)d_in[18];
  const float* Wframe  = (const float*)d_in[19];
  const float* bframe  = (const float*)d_in[20];
  const float* Wstop   = (const float*)d_in[21];
  const float* bstop   = (const float*)d_in[22];

  char* w = (char*)d_ws;
  size_t off = 0;
  auto take = [&](size_t bytes) -> void* {
    void* p = w + off;
    off = (off + bytes + 255) & ~(size_t)255;
    return p;
  };
  unsigned* barrier = (unsigned*)take(256);  // [cnt, gen]
  bf16* wattp   = (bf16*)take((size_t)(cKATT / 32) * (cNG / 16) * 512 * 2);
  bf16* wgenp   = (bf16*)take((size_t)(cKGEN / 32) * (cNG / 16) * 512 * 2);
  bf16* wqp     = (bf16*)take((size_t)(cH / 32) * (cA / 16) * 512 * 2);
  bf16* wfsp    = (bf16*)take((size_t)(1536 / 32) * 6 * 512 * 2);
  bf16* w1p     = (bf16*)take((size_t)3 * 16 * 512 * 2);
  bf16* w2p     = (bf16*)take((size_t)8 * 16 * 512 * 2);
  bf16* wmemp   = (bf16*)take((size_t)16 * 8 * 512 * 2);
  bf16* encb    = (bf16*)take((size_t)cB * cL * cE * 2);
  bf16* decin   = (bf16*)take((size_t)cT * cB * 96 * 2);
  bf16* pre1    = (bf16*)take((size_t)cT * cB * cP * 2);
  bf16* pre     = (bf16*)take((size_t)cT * cB * cP * 2);
  float* memproj = (float*)take((size_t)cB * cL * cA * 4);
  bf16* xatt    = (bf16*)take((size_t)cB * cKATT * 2);
  bf16* xgen    = (bf16*)take((size_t)cB * cKGEN * 2);
  float* catt   = (float*)take((size_t)cB * cH * 4);
  float* cgen   = (float*)take((size_t)cB * cH * 4);
  float* qbuf   = (float*)take((size_t)cB * cA * 4);
  float* feat   = (float*)take((size_t)cB * cFLT * cL * 4);
  float* prevw  = (float*)take((size_t)cB * cL * 4);
  float* cumw   = (float*)take((size_t)cB * cL * 4);

  float* out_spec  = (float*)d_out;                       // [B,T,OUT]
  float* out_stop  = out_spec + (size_t)cB * cT * cOUT;   // [B,T]
  float* out_align = out_stop + (size_t)cB * cT;          // [B,T,L]

  // ---- one-time prep: pack weights to bf16 B-fragments, prenet, mem_proj --
  pack_b_kernel<<<256, 256, 0, stream>>>(wattp, att_Wih, att_Whh, PK_ATT,
                                         cKATT / 32, cNG / 16);
  pack_b_kernel<<<256, 256, 0, stream>>>(wgenp, gen_Wih, gen_Whh, PK_GEN,
                                         cKGEN / 32, cNG / 16);
  pack_b_kernel<<<64, 256, 0, stream>>>(wqp, Wq, nullptr, PK_WQ, cH / 32,
                                        cA / 16);
  pack_b_kernel<<<64, 256, 0, stream>>>(wfsp, Wframe, Wstop, PK_WFS,
                                        1536 / 32, 6);
  pack_b_kernel<<<16, 256, 0, stream>>>(w1p, pre_W1, nullptr, PK_W1, 3, 16);
  pack_b_kernel<<<32, 256, 0, stream>>>(w2p, pre_W2, nullptr, PK_W2, 8, 16);
  pack_b_kernel<<<32, 256, 0, stream>>>(wmemp, Wmem, nullptr, PK_WMEM, 16, 8);
  cvt_enc_kernel<<<256, 256, 0, stream>>>(encb, enc, cB * cL * cE);
  build_decin_kernel<<<256, 256, 0, stream>>>(decin, target);

  // mem_proj [4096 x 128] = encb @ Wmem^T
  gemm_prep_kernel<<<(256 * 8 * 32 + TPB - 1) / TPB, TPB, 0, stream>>>(
      encb, cE, wmemp, 256, 8, 16, nullptr, 0, memproj, nullptr, cA);
  // prenet1 [4000 x 256] = relu(decin @ W1^T + b1)
  gemm_prep_kernel<<<(250 * 16 * 32 + TPB - 1) / TPB, TPB, 0, stream>>>(
      decin, 96, w1p, 250, 16, 3, pre_b1, 1, nullptr, pre1, cP);
  // prenet2 [4000 x 256] = relu(pre1 @ W2^T + b2)
  gemm_prep_kernel<<<(250 * 16 * 32 + TPB - 1) / TPB, TPB, 0, stream>>>(
      pre1, cP, w2p, 250, 16, 8, pre_b2, 1, nullptr, pre, cP);

  hipMemsetAsync(barrier, 0, 64, stream);

  tacotron_decode_kernel<<<NWG, TPB, 0, stream>>>(
      wattp, wgenp, wqp, wfsp, pre, memproj, enc, enclen, att_b, gen_b,
      bframe, bstop, Wconv, Wloc, vvec, xatt, xgen, catt, cgen, qbuf, feat,
      prevw, cumw, barrier, (volatile unsigned*)(barrier + 1), out_spec,
      out_stop, out_align);
}
